// MwAN_13941463843621
// MI455X (gfx1250) — compile-verified, matrix operand order checked by reading, not run on hardware
//
#include <hip/hip_runtime.h>
#include <hip/hip_bf16.h>

typedef __attribute__((ext_vector_type(16))) _Float16 v16h;
typedef __attribute__((ext_vector_type(8)))  float    v8f;

#define DEV static __device__ __forceinline__

DEV float sigm(float x) { return 1.0f / (1.0f + expf(-x)); }

// Load a 16-wide f16 K-fragment for WMMA f16 16x16x32 from a row-major f32 matrix.
// Per ISA 7.12.2 (16-bit A 16x32): lane m = lane&15, half = lane>>4;
// element e -> k = kb + 16*(e>>3) + 8*half + (e&7)  => two contiguous 8-float runs.
DEV v16h frag_rr(const float* __restrict__ base, int ld, int row, int kb, int half)
{
    const float* p0 = base + (size_t)row * ld + kb + 8 * half;
    v16h f;
#pragma unroll
    for (int i = 0; i < 8; i++) f[i]     = (_Float16)p0[i];
#pragma unroll
    for (int i = 0; i < 8; i++) f[8 + i] = (_Float16)p0[16 + i];
    return f;
}

DEV v8f wmma_f16(v16h a, v16h b, v8f c)
{
    return __builtin_amdgcn_wmma_f32_16x16x32_f16(false, a, false, b, (short)0, c,
                                                  false, false);
}

// ---------------------------------------------------------------------------
// Generic batched GEMM: C[m,n] = sum_k A[m,k] * W[n,k] (+ bias[n])
// A: (M x K) row-major ld=lda, W: (N x K) row-major ld=ldw. K % 32 == 0,
// M % 32 == 0, N % 32 == 0.
// grid = (M/32, N/32, batch), block = 32: one wave owns a 32x32 C tile
// (2x2 WMMA sub-tiles) so each fragment is reused twice -> 4 WMMAs per
// K-chunk and half the cache traffic per FLOP of a 16x16 tiling.
// ---------------------------------------------------------------------------
__global__ void k_gemm_rr(const float* __restrict__ A, int lda, long sA,
                          const float* __restrict__ W, int ldw, long sW,
                          const float* __restrict__ bias,
                          float* __restrict__ C, int ldc, long sC, int K)
{
    int lane = threadIdx.x;
    int half = lane >> 4, mn = lane & 15;
    const float* Ab = A + (size_t)blockIdx.z * sA;
    const float* Wb = W + (size_t)blockIdx.z * sW;
    float*       Cb = C + (size_t)blockIdx.z * sC;
    int row0 = blockIdx.x * 32;
    int col0 = blockIdx.y * 32;
    v8f acc[2][2] = {};
    for (int kb = 0; kb < K; kb += 32) {
        v16h a0 = frag_rr(Ab, lda, row0 + mn,      kb, half);
        v16h a1 = frag_rr(Ab, lda, row0 + 16 + mn, kb, half);
        v16h w0 = frag_rr(Wb, ldw, col0 + mn,      kb, half);
        v16h w1 = frag_rr(Wb, ldw, col0 + 16 + mn, kb, half);
        acc[0][0] = wmma_f16(a0, w0, acc[0][0]);
        acc[0][1] = wmma_f16(a0, w1, acc[0][1]);
        acc[1][0] = wmma_f16(a1, w0, acc[1][0]);
        acc[1][1] = wmma_f16(a1, w1, acc[1][1]);
    }
#pragma unroll
    for (int ni = 0; ni < 2; ni++) {
        int col = col0 + ni * 16 + mn;
        float bv = bias ? bias[col] : 0.0f;
#pragma unroll
        for (int mi = 0; mi < 2; mi++) {
#pragma unroll
            for (int v = 0; v < 8; v++) {
                // C/D layout: lane n = lane&15, VGPR v -> M = v (+8 for upper half)
                int m = row0 + mi * 16 + v + 8 * half;
                Cb[(size_t)m * ldc + col] = acc[mi][ni][v] + bv;
            }
        }
    }
}

// ---------------------------------------------------------------------------
// Pair attention scores (dot & self):
//   S[b,c,t] = sum_h tanh( sum_k A[b,t,k] * Hc[b,c,k] * W[h,k] ) * V[h]
// Per (b,c): 160x128x256 GEMM with rescaled weight (W ⊙ hc[b,c]).
// grid = (T/16, B*T), block = 32. A-frags for one 16-row tile live in VGPRs
// across all 8 N-tiles; hc[b,c] staged in LDS.
// ---------------------------------------------------------------------------
__global__ void k_pair_score(const float* __restrict__ A, const float* __restrict__ Hc,
                             const float* __restrict__ W, const float* __restrict__ V,
                             float* __restrict__ S, int T)
{
    int bc = blockIdx.y;
    int b  = bc / T;
    int t0 = blockIdx.x * 16;
    int lane = threadIdx.x;
    int half = lane >> 4, mn = lane & 15;

    __shared__ float hcs[256];
    const float* hv = Hc + (size_t)bc * 256;
    for (int i = lane; i < 256; i += 32) hcs[i] = hv[i];
    __syncthreads();

    const float* Ab = A + (size_t)b * T * 256;
    v16h af[8];
#pragma unroll
    for (int kc = 0; kc < 8; kc++) af[kc] = frag_rr(Ab, 256, t0 + mn, kc * 32, half);

    float part[8];
#pragma unroll
    for (int v = 0; v < 8; v++) part[v] = 0.0f;

    for (int nt = 0; nt < 8; nt++) {
        int n = nt * 16 + mn;
        const float* wr = W + (size_t)n * 256;
        v8f acc = {};
#pragma unroll
        for (int kc = 0; kc < 8; kc++) {
            const float* w0 = wr  + kc * 32 + 8 * half;
            const float* h0 = hcs + kc * 32 + 8 * half;
            v16h wf;
#pragma unroll
            for (int i = 0; i < 8; i++) wf[i]     = (_Float16)(w0[i]      * h0[i]);
#pragma unroll
            for (int i = 0; i < 8; i++) wf[8 + i] = (_Float16)(w0[16 + i] * h0[16 + i]);
            acc = wmma_f16(af[kc], wf, acc);
        }
        float vn = V[n];
#pragma unroll
        for (int v = 0; v < 8; v++) {
            float x = tanhf(acc[v]) * vn;
            // reduce across the 16 lanes sharing row m = v + 8*half
            x += __shfl_xor(x, 1, 32);
            x += __shfl_xor(x, 2, 32);
            x += __shfl_xor(x, 4, 32);
            x += __shfl_xor(x, 8, 32);
            part[v] += x;
        }
    }
    if (mn == 0) {
#pragma unroll
        for (int v = 0; v < 8; v++)
            S[(size_t)bc * T + t0 + v + 8 * half] = part[v];
    }
}

// Elementwise (factored) scores for concat/minus attention:
//   S[b,c,t] = sum_h tanh(Pt[b,t,h] + sign*Pc[b,c,h]) * V[h]
__global__ void k_ew_score(const float* __restrict__ Pt, const float* __restrict__ Pc,
                           const float* __restrict__ V, float sign,
                           float* __restrict__ S, int T)
{
    long idx = (long)blockIdx.x * blockDim.x + threadIdx.x;
    long total = (long)8 * T * T;
    if (idx >= total) return;
    int t  = (int)(idx % T);
    long bc = idx / T;
    int c  = (int)(bc % T);
    int b  = (int)(bc / T);
    const float* pt = Pt + ((size_t)b * T + t) * 128;
    const float* pc = Pc + ((size_t)b * T + c) * 128;
    float s = 0.0f;
#pragma unroll 4
    for (int h = 0; h < 128; h++) s += tanhf(pt[h] + sign * pc[h]) * V[h];
    S[idx] = s;
}

// Row softmax (in place), one wave per row of length L.
__global__ void k_softmax_rows(float* __restrict__ X, int L)
{
    int lane = threadIdx.x;
    float* x = X + (size_t)blockIdx.x * L;
    float mx = -1e30f;
    for (int i = lane; i < L; i += 32) mx = fmaxf(mx, x[i]);
    for (int m = 1; m < 32; m <<= 1) mx = fmaxf(mx, __shfl_xor(mx, m, 32));
    float sum = 0.0f;
    for (int i = lane; i < L; i += 32) { float e = expf(x[i] - mx); x[i] = e; sum += e; }
    for (int m = 1; m < 32; m <<= 1) sum += __shfl_xor(sum, m, 32);
    float inv = 1.0f / sum;
    for (int i = lane; i < L; i += 32) x[i] *= inv;
}

// GRU recurrence. One block per (batch, direction-sequence). 384 threads.
// xg: (B,T,384) with bih already folded in. out column base chosen by caller.
__global__ void k_gru_scan(const float* __restrict__ xg,
                           const float* __restrict__ Whh, const float* __restrict__ bhh,
                           float* __restrict__ out, int ldo, int T, int reverse)
{
    int b = blockIdx.x;
    int g = threadIdx.x;          // 0..383
    __shared__ float hs[128];
    __shared__ float gh[384];
    if (g < 128) hs[g] = 0.0f;
    __syncthreads();
    const float* wr = Whh + (size_t)g * 128;
    float bb = bhh[g];
    for (int step = 0; step < T; step++) {
        int t = reverse ? (T - 1 - step) : step;
        float acc = bb;
#pragma unroll 8
        for (int k = 0; k < 128; k++) acc += wr[k] * hs[k];
        gh[g] = acc;
        __syncthreads();
        if (g < 128) {
            const float* xt = xg + ((size_t)b * T + t) * 384;
            float r = sigm(xt[g]       + gh[g]);
            float z = sigm(xt[128 + g] + gh[128 + g]);
            float n = tanhf(xt[256 + g] + r * gh[256 + g]);
            float h = (1.0f - z) * n + z * hs[g];
            hs[g] = h;
            out[((size_t)b * T + t) * ldo + g] = h;
        }
        __syncthreads();
    }
}

// Embedding gather with zero pad 300 -> 320 columns.
__global__ void k_gather_pad(const int* __restrict__ text, const float* __restrict__ emb,
                             float* __restrict__ out, long rows)
{
    long idx = (long)blockIdx.x * blockDim.x + threadIdx.x;
    long total = rows * 320;
    if (idx >= total) return;
    int col = (int)(idx % 320);
    long row = idx / 320;
    int tok = text[row];
    out[idx] = (col < 300) ? emb[(size_t)tok * 300 + col] : 0.0f;
}

// Zero-pad weight rows incols -> outcols.
__global__ void k_pad_rows(const float* __restrict__ in, float* __restrict__ out,
                           int incols, int outcols, long rows)
{
    long idx = (long)blockIdx.x * blockDim.x + threadIdx.x;
    long total = rows * outcols;
    if (idx >= total) return;
    int col = (int)(idx % outcols);
    long row = idx / outcols;
    out[idx] = (col < incols) ? in[row * incols + col] : 0.0f;
}

// (B,T,D) -> (B,D,T)
__global__ void k_transpose(const float* __restrict__ in, float* __restrict__ out,
                            int T, int D)
{
    long idx = (long)blockIdx.x * blockDim.x + threadIdx.x;
    long total = (long)8 * T * D;
    if (idx >= total) return;
    int d = (int)(idx % D);
    long bt = idx / D;
    int t = (int)(bt % T);
    int b = (int)(bt / T);
    out[((size_t)b * D + d) * T + t] = in[idx];
}

// Copy (rows x cols), src ld = cols, dst ld = dstld.
__global__ void k_copy_stride(const float* __restrict__ src, float* __restrict__ dst,
                              int cols, int dstld, long rows)
{
    long idx = (long)blockIdx.x * blockDim.x + threadIdx.x;
    long total = rows * cols;
    if (idx >= total) return;
    int c = (int)(idx % cols);
    long r = idx / cols;
    dst[r * dstld + c] = src[idx];
}

// out[i] = sum_h tanh(P[i,h]) * V[h]
__global__ void k_tanh_dot(const float* __restrict__ P, const float* __restrict__ V,
                           float* __restrict__ out, int n, int hd)
{
    int idx = blockIdx.x * blockDim.x + threadIdx.x;
    if (idx >= n) return;
    const float* p = P + (size_t)idx * hd;
    float s = 0.0f;
    for (int h = 0; h < hd; h++) s += tanhf(p[h]) * V[h];
    out[idx] = s;
}

// out[b,d] = sum_t w[b,t] * X[b,t,d]
__global__ void k_weighted_sum(const float* __restrict__ w, const float* __restrict__ X,
                               float* __restrict__ out, int L, int D)
{
    int idx = blockIdx.x * blockDim.x + threadIdx.x;
    if (idx >= 8 * D) return;
    int b = idx / D, d = idx % D;
    float s = 0.0f;
    for (int t = 0; t < L; t++) s += w[b * L + t] * X[((size_t)b * L + t) * D + d];
    out[idx] = s;
}

// out[k] = sum_h W[h,k] * V[h]   (W: 128x256)
__global__ void k_project_v(const float* __restrict__ W, const float* __restrict__ V,
                            float* __restrict__ out)
{
    int k = threadIdx.x;
    if (k >= 256) return;
    float s = 0.0f;
    for (int h = 0; h < 128; h++) s += W[h * 256 + k] * V[h];
    out[k] = s;
}

// sc[b,c] = agg[b,c,:]·w1v + rp[b,:]·w2v   (linear factoring of (·)@Vv)
__global__ void k_sc(const float* __restrict__ agg, const float* __restrict__ rp,
                     const float* __restrict__ w1v, const float* __restrict__ w2v,
                     float* __restrict__ sc)
{
    int idx = blockIdx.x * blockDim.x + threadIdx.x;
    if (idx >= 1280) return;
    int b = idx / 160;
    const float* a = agg + (size_t)idx * 256;
    const float* r = rp + b * 256;
    float s = 0.0f;
    for (int k = 0; k < 256; k++) s += a[k] * w1v[k] + r[k] * w2v[k];
    sc[idx] = s;
}

// out[b,o] = rc[b,:]·Wout[o,:] + bout[o]
__global__ void k_out(const float* __restrict__ rc, const float* __restrict__ Wout,
                      const float* __restrict__ bout, float* __restrict__ out)
{
    int idx = threadIdx.x;
    if (idx >= 16) return;
    int b = idx >> 1, o = idx & 1;
    float s = bout[o];
    for (int d = 0; d < 256; d++) s += rc[b * 256 + d] * Wout[o * 256 + d];
    out[idx] = s;
}

// ---------------------------------------------------------------------------
static inline void gemm(hipStream_t s, const float* A, int lda, long sA,
                        const float* W, int ldw, long sW, const float* bias,
                        float* C, int ldc, long sC, int M, int N, int K, int batch)
{
    dim3 g(M / 32, N / 32, batch);
    k_gemm_rr<<<g, dim3(32), 0, s>>>(A, lda, sA, W, ldw, sW, bias, C, ldc, sC, K);
}

extern "C" void kernel_launch(void* const* d_in, const int* in_sizes, int n_in,
                              void* d_out, int out_size, void* d_ws, size_t ws_size,
                              hipStream_t stream)
{
    const int B = 8, T = 160, G = 384, EP = 320, AIN = 1536;
    const int BT = B * T;                  // 1280
    (void)in_sizes; (void)n_in; (void)out_size;

    const int*   text_a = (const int*)d_in[0];
    const int*   text_b = (const int*)d_in[1];
    const float* emb    = (const float*)d_in[2];
    const float* pWihF = (const float*)d_in[3],  *pWhhF = (const float*)d_in[4];
    const float* pBihF = (const float*)d_in[5],  *pBhhF = (const float*)d_in[6];
    const float* pWihB = (const float*)d_in[7],  *pWhhB = (const float*)d_in[8];
    const float* pBihB = (const float*)d_in[9],  *pBhhB = (const float*)d_in[10];
    const float* cWihF = (const float*)d_in[11], *cWhhF = (const float*)d_in[12];
    const float* cBihF = (const float*)d_in[13], *cBhhF = (const float*)d_in[14];
    const float* cWihB = (const float*)d_in[15], *cWhhB = (const float*)d_in[16];
    const float* cBihB = (const float*)d_in[17], *cBhhB = (const float*)d_in[18];
    const float* aWihF = (const float*)d_in[19], *aWhhF = (const float*)d_in[20];
    const float* aBihF = (const float*)d_in[21], *aBhhF = (const float*)d_in[22];
    const float* aWihB = (const float*)d_in[23], *aWhhB = (const float*)d_in[24];
    const float* aBihB = (const float*)d_in[25], *aBhhB = (const float*)d_in[26];
    const float* Wc1 = (const float*)d_in[27], *Wc2 = (const float*)d_in[28];
    const float* Vc  = (const float*)d_in[29];
    const float* Wb  = (const float*)d_in[30];
    const float* Wd  = (const float*)d_in[31], *Vd = (const float*)d_in[32];
    const float* Wm  = (const float*)d_in[33], *Vm = (const float*)d_in[34];
    const float* Wsw = (const float*)d_in[35], *Vs = (const float*)d_in[36];
    const float* Wp  = (const float*)d_in[37], *Vp = (const float*)d_in[38];
    const float* W1  = (const float*)d_in[39], *W2 = (const float*)d_in[40];
    const float* Vv  = (const float*)d_in[41];
    const float* Wout = (const float*)d_in[42], *bout = (const float*)d_in[43];

    float* ws = (float*)d_ws;
    size_t off = 0;
    auto take = [&](size_t n) { float* p = ws + off; off += n; return p; };

    float* xa    = take((size_t)BT * EP);       // gathered emb[text_a], padded
    float* xb    = take((size_t)BT * EP);
    float* wpPF  = take((size_t)G * EP);        // padded Wih (E 300 -> 320)
    float* wpPB  = take((size_t)G * EP);
    float* wpCF  = take((size_t)G * EP);
    float* wpCB  = take((size_t)G * EP);
    float* xgPF  = take((size_t)BT * G);
    float* xgPB  = take((size_t)BT * G);
    float* xgCF  = take((size_t)BT * G);
    float* xgCB  = take((size_t)BT * G);
    float* hp    = take((size_t)BT * 256);
    float* hc    = take((size_t)BT * 256);
    float* hpT   = take((size_t)BT * 256);      // (B,256,T)
    float* hcT   = take((size_t)BT * 256);
    float* Pp1   = take((size_t)BT * 128);      // hp @ Wc1^T
    float* Pc2   = take((size_t)BT * 128);      // hc @ Wc2^T
    float* Pmp   = take((size_t)BT * 128);      // hp @ Wm^T
    float* Pmc   = take((size_t)BT * 128);      // hc @ Wm^T
    float* Ppp   = take((size_t)BT * 128);      // hp @ Wp^T
    float* Pb    = take((size_t)BT * 256);      // hp @ Wb^T
    float* attn  = take((size_t)B * T * T);     // reused for all 5 attentions
    float* aggin = take((size_t)BT * AIN);
    float* xgAF  = take((size_t)BT * G);
    float* xgAB  = take((size_t)BT * G);
    float* agg   = take((size_t)BT * 256);
    float* scp   = take((size_t)BT);            // ap logits/weights
    float* rp    = take((size_t)B * 256);
    float* w1v   = take((size_t)256);
    float* w2v   = take((size_t)256);
    float* sc    = take((size_t)BT);
    float* rc    = take((size_t)B * 256);
    if (off * sizeof(float) > ws_size) return;  // workspace insufficient

    dim3 blk(256);
    auto g1 = [](long n) { return dim3((unsigned)((n + 255) / 256)); };

    // --- Embedding gather + weight padding -------------------------------
    k_gather_pad<<<g1((long)BT * EP), blk, 0, stream>>>(text_a, emb, xa, BT);
    k_gather_pad<<<g1((long)BT * EP), blk, 0, stream>>>(text_b, emb, xb, BT);
    k_pad_rows<<<g1((long)G * EP), blk, 0, stream>>>(pWihF, wpPF, 300, EP, G);
    k_pad_rows<<<g1((long)G * EP), blk, 0, stream>>>(pWihB, wpPB, 300, EP, G);
    k_pad_rows<<<g1((long)G * EP), blk, 0, stream>>>(cWihF, wpCF, 300, EP, G);
    k_pad_rows<<<g1((long)G * EP), blk, 0, stream>>>(cWihB, wpCB, 300, EP, G);

    // --- Input-gate GEMMs (WMMA): xg = x @ Wih^T + bih -------------------
    gemm(stream, xa, EP, 0, wpPF, EP, 0, pBihF, xgPF, G, 0, BT, G, EP, 1);
    gemm(stream, xa, EP, 0, wpPB, EP, 0, pBihB, xgPB, G, 0, BT, G, EP, 1);
    gemm(stream, xb, EP, 0, wpCF, EP, 0, cBihF, xgCF, G, 0, BT, G, EP, 1);
    gemm(stream, xb, EP, 0, wpCB, EP, 0, cBihB, xgCB, G, 0, BT, G, EP, 1);

    // --- BiGRU scans (latency-bound recurrences) -------------------------
    k_gru_scan<<<dim3(B), dim3(G), 0, stream>>>(xgPF, pWhhF, pBhhF, hp,       256, T, 0);
    k_gru_scan<<<dim3(B), dim3(G), 0, stream>>>(xgPB, pWhhB, pBhhB, hp + 128, 256, T, 1);
    k_gru_scan<<<dim3(B), dim3(G), 0, stream>>>(xgCF, cWhhF, cBhhF, hc,       256, T, 0);
    k_gru_scan<<<dim3(B), dim3(G), 0, stream>>>(xgCB, cWhhB, cBhhB, hc + 128, 256, T, 1);

    // --- Projections (WMMA) ----------------------------------------------
    gemm(stream, hp, 256, 0, Wc1, 256, 0, nullptr, Pp1, 128, 0, BT, 128, 256, 1);
    gemm(stream, hc, 256, 0, Wc2, 256, 0, nullptr, Pc2, 128, 0, BT, 128, 256, 1);
    gemm(stream, hp, 256, 0, Wm,  256, 0, nullptr, Pmp, 128, 0, BT, 128, 256, 1);
    gemm(stream, hc, 256, 0, Wm,  256, 0, nullptr, Pmc, 128, 0, BT, 128, 256, 1);
    gemm(stream, hp, 256, 0, Wp,  256, 0, nullptr, Ppp, 128, 0, BT, 128, 256, 1);
    gemm(stream, hp, 256, 0, Wb,  256, 0, nullptr, Pb,  256, 0, BT, 256, 256, 1);
    k_transpose<<<g1((long)BT * 256), blk, 0, stream>>>(hp, hpT, T, 256);
    k_transpose<<<g1((long)BT * 256), blk, 0, stream>>>(hc, hcT, T, 256);

    const long sAttn = (long)T * T, sHT = (long)256 * T, sH = (long)T * 256;

    // agg_in = [hc | self | concat | dot | bil | minus]
    k_copy_stride<<<g1((long)BT * 256), blk, 0, stream>>>(hc, aggin, 256, AIN, BT);

    // --- concat attention: factored tanh(Pp1 + Pc2)·Vc -------------------
    k_ew_score<<<g1((long)B * T * T), blk, 0, stream>>>(Pp1, Pc2, Vc, 1.0f, attn, T);
    k_softmax_rows<<<dim3(BT), dim3(32), 0, stream>>>(attn, T);
    gemm(stream, attn, T, sAttn, hpT, T, sHT, nullptr, aggin + 512, AIN, (long)T * AIN,
         T, 256, T, B);

    // --- bilinear attention: logits = hc @ (hp Wb^T)^T (batched WMMA) ----
    gemm(stream, hc, 256, sH, Pb, 256, sH, nullptr, attn, T, sAttn, T, T, 256, B);
    k_softmax_rows<<<dim3(BT), dim3(32), 0, stream>>>(attn, T);
    gemm(stream, attn, T, sAttn, hpT, T, sHT, nullptr, aggin + 1024, AIN, (long)T * AIN,
         T, 256, T, B);

    // --- dot attention: per-(b,c) WMMA GEMM with Wd ⊙ hc[b,c] ------------
    k_pair_score<<<dim3(T / 16, BT), dim3(32), 0, stream>>>(hp, hc, Wd, Vd, attn, T);
    k_softmax_rows<<<dim3(BT), dim3(32), 0, stream>>>(attn, T);
    gemm(stream, attn, T, sAttn, hpT, T, sHT, nullptr, aggin + 768, AIN, (long)T * AIN,
         T, 256, T, B);

    // --- minus attention: factored tanh(Pmp - Pmc)·Vm --------------------
    k_ew_score<<<g1((long)B * T * T), blk, 0, stream>>>(Pmp, Pmc, Vm, -1.0f, attn, T);
    k_softmax_rows<<<dim3(BT), dim3(32), 0, stream>>>(attn, T);
    gemm(stream, attn, T, sAttn, hpT, T, sHT, nullptr, aggin + 1280, AIN, (long)T * AIN,
         T, 256, T, B);

    // --- self attention: per-(b,c) WMMA GEMM with Ws ⊙ hc[b,c] -----------
    k_pair_score<<<dim3(T / 16, BT), dim3(32), 0, stream>>>(hc, hc, Wsw, Vs, attn, T);
    k_softmax_rows<<<dim3(BT), dim3(32), 0, stream>>>(attn, T);
    gemm(stream, attn, T, sAttn, hcT, T, sHT, nullptr, aggin + 256, AIN, (long)T * AIN,
         T, 256, T, B);

    // --- aggregation BiGRU ------------------------------------------------
    gemm(stream, aggin, AIN, 0, aWihF, AIN, 0, aBihF, xgAF, G, 0, BT, G, AIN, 1);
    gemm(stream, aggin, AIN, 0, aWihB, AIN, 0, aBihB, xgAB, G, 0, BT, G, AIN, 1);
    k_gru_scan<<<dim3(B), dim3(G), 0, stream>>>(xgAF, aWhhF, aBhhF, agg,       256, T, 0);
    k_gru_scan<<<dim3(B), dim3(G), 0, stream>>>(xgAB, aWhhB, aBhhB, agg + 128, 256, T, 1);

    // --- prediction head --------------------------------------------------
    k_tanh_dot<<<g1(BT), blk, 0, stream>>>(Ppp, Vp, scp, BT, 128);
    k_softmax_rows<<<dim3(B), dim3(32), 0, stream>>>(scp, T);
    k_weighted_sum<<<g1(8 * 256), blk, 0, stream>>>(scp, hp, rp, T, 256);
    k_project_v<<<dim3(1), blk, 0, stream>>>(W1, Vv, w1v);
    k_project_v<<<dim3(1), blk, 0, stream>>>(W2, Vv, w2v);
    k_sc<<<g1(BT), blk, 0, stream>>>(agg, rp, w1v, w2v, sc);
    k_softmax_rows<<<dim3(B), dim3(32), 0, stream>>>(sc, T);
    k_weighted_sum<<<g1(8 * 256), blk, 0, stream>>>(sc, agg, rc, T, 256);
    k_out<<<dim3(1), dim3(32), 0, stream>>>(rc, Wout, bout, (float*)d_out);
}
